// SelfAttention_2327872274725
// MI455X (gfx1250) — matrix operand order repaired and checked by reading, not compiled, not run on hardware
//
#include <hip/hip_runtime.h>

#define BATCH 8
#define CIN 128
#define NPIX (192 * 192)          // 36864
#define GRAM_WGS 32               // workgroups per batch for the Gram pass
#define GRAM_ITERS (NPIX / (32 * GRAM_WGS))   // 36 chunks of 32 pixels
#define FIN_WGS 72                // workgroups per batch for the final pass
#define FIN_ITERS (NPIX / (32 * FIN_WGS))     // 16 chunks of 32 pixels

typedef __attribute__((ext_vector_type(16))) __bf16 v16bf;
typedef __attribute__((ext_vector_type(8)))  float  v8f;

union FragAB { uint4 q[2]; v16bf v; };

// round-to-nearest-even f32 -> bf16 bits
__device__ __forceinline__ unsigned f2bf(float f) {
  unsigned u = __float_as_uint(f);
  return ((u + 0x7FFFu + ((u >> 16) & 1u)) >> 16) & 0xFFFFu;
}

// ---------------------------------------------------------------------------
// Kernel 1: per-workgroup partial Gram  S_part = X_chunk * X_chunk^T  (bf16 WMMA)
// plus partial row sums s_part = X_chunk * 1.   x layout: (B, C=128, N=36864)
// ---------------------------------------------------------------------------
__global__ __launch_bounds__(256) void k_gram(const float* __restrict__ x,
                                              float* __restrict__ pS,
                                              float* __restrict__ pSum)
{
  const int wg   = blockIdx.x;
  const int b    = wg / GRAM_WGS;
  const int sub  = wg % GRAM_WGS;
  const int tid  = threadIdx.x;
  const int lane = tid & 31;
  const int wave = tid >> 5;          // 0..7 -> d-block
  const int l15  = lane & 15;
  const int hi   = lane >> 4;         // 0/1

  // bf16 chunk: 128 channels x 32 pixels, row stride 40 halves (80B = 20 dwords)
  __shared__ __align__(16) unsigned short xs[128 * 40];

  const float* xg = x + (size_t)b * CIN * NPIX;

  const int d0  = wave * 16;
  const int aO0 = (d0 + l15) * 40 + hi * 8;   // A frag: K {0-7 | 8-15}
  const int aO1 = aO0 + 16;                   //         K {16-23 | 24-31}
  const int bBase = l15 * 40 + hi * 8;        // B frag: K {0-7,16-23 | 8-15,24-31} of row l15

  v8f acc[8];
#pragma unroll
  for (int e = 0; e < 8; ++e) acc[e] = (v8f){0.f,0.f,0.f,0.f,0.f,0.f,0.f,0.f};
  float sacc[4] = {0.f, 0.f, 0.f, 0.f};

  const int nbase = sub * (GRAM_ITERS * 32);
  for (int it = 0; it < GRAM_ITERS; ++it) {
    const int n0 = nbase + it * 32;
    __syncthreads();
#pragma unroll
    for (int i = 0; i < 4; ++i) {           // 256 thr * 4 = 128 rows * 8 float4
      const int flat = i * 256 + tid;
      const int c  = flat >> 3;
      const int f4 = flat & 7;
      const float4 v = *(const float4*)(xg + (size_t)c * NPIX + n0 + f4 * 4);
      sacc[i] += v.x + v.y + v.z + v.w;
      uint2 p;
      p.x = f2bf(v.x) | (f2bf(v.y) << 16);
      p.y = f2bf(v.z) | (f2bf(v.w) << 16);
      *(uint2*)(xs + c * 40 + f4 * 4) = p;
      if (it + 1 < GRAM_ITERS)
        __builtin_prefetch(xg + (size_t)c * NPIX + n0 + 32 + f4 * 4, 0, 0);
    }
    __syncthreads();
    FragAB a;
    a.q[0] = *(const uint4*)(xs + aO0);
    a.q[1] = *(const uint4*)(xs + aO1);
#pragma unroll
    for (int e = 0; e < 8; ++e) {
      FragAB bb;
      const int ro = e * 16 * 40 + bBase;
      bb.q[0] = *(const uint4*)(xs + ro);
      bb.q[1] = *(const uint4*)(xs + ro + 16);
      acc[e] = __builtin_amdgcn_wmma_f32_16x16x32_bf16(
          false, a.v, false, bb.v, (short)0, acc[e], false, false);
    }
  }

  // D layout: lanes<16 -> M = r, lanes>=16 -> M = 8+r; N = lane%16
#pragma unroll
  for (int e = 0; e < 8; ++e) {
#pragma unroll
    for (int r = 0; r < 8; ++r) {
      const int row = d0 + r + hi * 8;
      const int col = e * 16 + l15;
      pS[((size_t)wg * 128 + row) * 128 + col] = acc[e][r];
    }
  }
  // partial row sums: each group of 8 consecutive lanes shares one channel
#pragma unroll
  for (int i = 0; i < 4; ++i) {
    float s = sacc[i];
    s += __shfl_xor(s, 1, 32);
    s += __shfl_xor(s, 2, 32);
    s += __shfl_xor(s, 4, 32);
    if ((tid & 7) == 0) {
      const int c = (i * 256 + tid) >> 3;
      pSum[(size_t)wg * 128 + c] = s;
    }
  }
}

// ---------------------------------------------------------------------------
// Kernel 2: deterministic reduction of partial Grams / row sums
// ---------------------------------------------------------------------------
__global__ __launch_bounds__(256) void k_reduce(const float* __restrict__ pS,
                                                const float* __restrict__ pSum,
                                                float* __restrict__ S,
                                                float* __restrict__ ssum)
{
  const int g = blockIdx.x * 256 + threadIdx.x;   // 0..32767
#pragma unroll
  for (int j = 0; j < 4; ++j) {
    const int idx = g + j * 32768;                // 8*16384 entries
    const int b  = idx >> 14;
    const int rc = idx & 16383;
    float s = 0.f;
    for (int p = 0; p < GRAM_WGS; ++p)
      s += pS[(((size_t)(b * GRAM_WGS + p)) << 14) + rc];
    S[idx] = s;
  }
  if (g < BATCH * 128) {
    const int b = g >> 7, c = g & 127;
    float s = 0.f;
    for (int p = 0; p < GRAM_WGS; ++p)
      s += pSum[(size_t)(b * GRAM_WGS + p) * 128 + c];
    ssum[g] = s;
  }
}

// ---------------------------------------------------------------------------
// Kernel 3: energy = Wq S Wk^T + (Wq s)bk^T + bq(Wk s)^T + N bq bk^T ; softmax
// One WG per (batch, 16-row block of d).
// ---------------------------------------------------------------------------
__global__ __launch_bounds__(256) void k_energy_softmax(
    const float* __restrict__ S, const float* __restrict__ ssum,
    const float* __restrict__ Wq, const float* __restrict__ bq,
    const float* __restrict__ Wk, const float* __restrict__ bk,
    float* __restrict__ A)
{
  const int b    = blockIdx.x >> 3;
  const int dblk = blockIdx.x & 7;
  const int d0   = dblk * 16;
  const int tid  = threadIdx.x;

  __shared__ float Ms[128 * 129];   // staged S, then staged Wk (pad 129 vs banks)
  __shared__ float T1s[16 * 132];
  __shared__ float tarr[128];
  __shared__ float uarr[16];

  const float* Sb = S + (size_t)b * 16384;
  const float* sb = ssum + b * 128;

  for (int k = 0; k < 64; ++k) {
    const int idx = k * 256 + tid;
    Ms[(idx >> 7) * 129 + (idx & 127)] = Sb[idx];
  }
  if (tid < 128) {                       // t[e] = Wk[e] . s
    float a = 0.f;
    for (int c = 0; c < 128; ++c) a += Wk[tid * 128 + c] * sb[c];
    tarr[tid] = a;
  } else if (tid < 144) {                // u[d] = Wq[d0+d] . s
    const int d = tid - 128;
    float a = 0.f;
    for (int c = 0; c < 128; ++c) a += Wq[(d0 + d) * 128 + c] * sb[c];
    uarr[d] = a;
  }
  __syncthreads();

  {  // T1 = Wq[d0:d0+16,:] * S   (16 x 128 into LDS)
    const int dr = tid >> 4;
    const int cb = (tid & 15) * 8;
    float acc[8] = {0.f,0.f,0.f,0.f,0.f,0.f,0.f,0.f};
    for (int c = 0; c < 128; ++c) {
      const float wq = Wq[(d0 + dr) * 128 + c];
      const float* Sr = &Ms[c * 129 + cb];
#pragma unroll
      for (int j = 0; j < 8; ++j) acc[j] += wq * Sr[j];
    }
#pragma unroll
    for (int j = 0; j < 8; ++j) T1s[dr * 132 + cb + j] = acc[j];
  }
  __syncthreads();

  for (int k = 0; k < 64; ++k) {         // restage with Wk
    const int idx = k * 256 + tid;
    Ms[(idx >> 7) * 129 + (idx & 127)] = Wk[idx];
  }
  __syncthreads();

  const int r  = tid >> 4;               // row in block
  const int e0 = (tid & 15) * 8;         // 8 columns per thread
  const float bqv = bq[d0 + r];
  const float uv  = uarr[r];
  float en[8];
#pragma unroll
  for (int j = 0; j < 8; ++j) {
    const int e = e0 + j;
    float a = 0.f;
    for (int c = 0; c < 128; ++c) a += T1s[r * 132 + c] * Ms[e * 129 + c];
    en[j] = a + bqv * tarr[e] + uv * bk[e] + 36864.0f * bqv * bk[e];
  }
  float m = en[0];
#pragma unroll
  for (int j = 1; j < 8; ++j) m = fmaxf(m, en[j]);
  m = fmaxf(m, __shfl_xor(m, 1, 32));
  m = fmaxf(m, __shfl_xor(m, 2, 32));
  m = fmaxf(m, __shfl_xor(m, 4, 32));
  m = fmaxf(m, __shfl_xor(m, 8, 32));
  float sum = 0.f;
#pragma unroll
  for (int j = 0; j < 8; ++j) { en[j] = __expf(en[j] - m); sum += en[j]; }
  sum += __shfl_xor(sum, 1, 32);
  sum += __shfl_xor(sum, 2, 32);
  sum += __shfl_xor(sum, 4, 32);
  sum += __shfl_xor(sum, 8, 32);
  const float inv = 1.0f / sum;
#pragma unroll
  for (int j = 0; j < 8; ++j)
    A[(size_t)b * 16384 + (d0 + r) * 128 + e0 + j] = en[j] * inv;
}

// ---------------------------------------------------------------------------
// Kernel 4: M = gamma * Wo * (A * Wv) stored as bf16;  c = gamma*(Wo*(A bv)+bo)
// One WG per (batch, 16-column block).
// ---------------------------------------------------------------------------
__global__ __launch_bounds__(256) void k_fold(
    const float* __restrict__ A, const float* __restrict__ Wv,
    const float* __restrict__ bv, const float* __restrict__ Wo,
    const float* __restrict__ bo, const float* __restrict__ gamma,
    unsigned short* __restrict__ Mbf, float* __restrict__ cvec)
{
  const int b    = blockIdx.x >> 3;
  const int cblk = blockIdx.x & 7;
  const int c0   = cblk * 16;
  const int tid  = threadIdx.x;

  __shared__ float Pblk[128 * 17];
  __shared__ float avbl[128];

  const float* Ab = A + (size_t)b * 16384;
  const float g = gamma[0];

  const int dd  = tid >> 1;            // row (d, then o)
  const int cc0 = (tid & 1) * 8;       // 8-column half of the 16-col block
  {
    float acc[8] = {0.f,0.f,0.f,0.f,0.f,0.f,0.f,0.f};
    for (int e = 0; e < 128; ++e) {
      const float a = Ab[dd * 128 + e];
      const float* wv = Wv + e * 128 + c0 + cc0;
#pragma unroll
      for (int j = 0; j < 8; ++j) acc[j] += a * wv[j];
    }
#pragma unroll
    for (int j = 0; j < 8; ++j) Pblk[dd * 17 + cc0 + j] = acc[j];
  }
  if (tid < 128) {                     // avb = A bv
    float s = 0.f;
    for (int e = 0; e < 128; ++e) s += Ab[tid * 128 + e] * bv[e];
    avbl[tid] = s;
  }
  __syncthreads();
  {
    float acc[8] = {0.f,0.f,0.f,0.f,0.f,0.f,0.f,0.f};
    for (int d = 0; d < 128; ++d) {
      const float wo = Wo[dd * 128 + d];
#pragma unroll
      for (int j = 0; j < 8; ++j) acc[j] += wo * Pblk[d * 17 + cc0 + j];
    }
    uint4 pk;
    pk.x = f2bf(g * acc[0]) | (f2bf(g * acc[1]) << 16);
    pk.y = f2bf(g * acc[2]) | (f2bf(g * acc[3]) << 16);
    pk.z = f2bf(g * acc[4]) | (f2bf(g * acc[5]) << 16);
    pk.w = f2bf(g * acc[6]) | (f2bf(g * acc[7]) << 16);
    *(uint4*)(Mbf + ((size_t)b * 128 + dd) * 128 + c0 + cc0) = pk;
  }
  if (cblk == 0 && tid < 128) {
    float s = 0.f;
    for (int d = 0; d < 128; ++d) s += Wo[tid * 128 + d] * avbl[d];
    cvec[b * 128 + tid] = g * (s + bo[tid]);
  }
}

// ---------------------------------------------------------------------------
// Kernel 5: out = M*x + c + x  (bf16 WMMA, M fragments resident in VGPRs)
// ---------------------------------------------------------------------------
__global__ __launch_bounds__(256) void k_final(
    const float* __restrict__ x, const unsigned short* __restrict__ Mbf,
    const float* __restrict__ cvec, float* __restrict__ out)
{
  const int wgid = blockIdx.x;
  const int b    = wgid / FIN_WGS;
  const int sub  = wgid % FIN_WGS;
  const int tid  = threadIdx.x;
  const int lane = tid & 31;
  const int wave = tid >> 5;
  const int l15  = lane & 15;
  const int hi   = lane >> 4;

  // transposed chunk: [pixel 0..31][channel 0..127], stride 136 halves (16B aligned)
  __shared__ __align__(16) unsigned short xs[32 * 136];

  const float* xg = x + (size_t)b * CIN * NPIX;
  const int ow = wave * 16;            // out-channel block of this wave

  FragAB m[4];                         // A fragments of M, reused all chunks
#pragma unroll
  for (int kb = 0; kb < 4; ++kb) {
    const size_t base = ((size_t)b * 128 + ow + l15) * 128 + kb * 32 + hi * 8;
    m[kb].q[0] = *(const uint4*)(Mbf + base);
    m[kb].q[1] = *(const uint4*)(Mbf + base + 16);
  }
  float cv[8];
#pragma unroll
  for (int r = 0; r < 8; ++r) cv[r] = cvec[b * 128 + ow + r + hi * 8];

  const int nbase = sub * (FIN_ITERS * 32);
  for (int it = 0; it < FIN_ITERS; ++it) {
    const int n0 = nbase + it * 32;
    __syncthreads();
#pragma unroll
    for (int i = 0; i < 4; ++i) {
      const int flat = i * 256 + tid;
      const int c  = flat >> 3;
      const int f4 = flat & 7;
      const float4 v = *(const float4*)(xg + (size_t)c * NPIX + n0 + f4 * 4);
      const int rr = f4 * 4;
      xs[(rr + 0) * 136 + c] = (unsigned short)f2bf(v.x);
      xs[(rr + 1) * 136 + c] = (unsigned short)f2bf(v.y);
      xs[(rr + 2) * 136 + c] = (unsigned short)f2bf(v.z);
      xs[(rr + 3) * 136 + c] = (unsigned short)f2bf(v.w);
      if (it + 1 < FIN_ITERS)
        __builtin_prefetch(xg + (size_t)c * NPIX + n0 + 32 + f4 * 4, 0, 0);
    }
    __syncthreads();
#pragma unroll
    for (int pt = 0; pt < 2; ++pt) {   // two 16-pixel tiles per chunk
      v8f acc = (v8f){0.f,0.f,0.f,0.f,0.f,0.f,0.f,0.f};
      const int boff = (pt * 16 + l15) * 136 + hi * 8;
#pragma unroll
      for (int kb = 0; kb < 4; ++kb) { // K = 128 in 4 steps of 32
        FragAB bb;
        bb.q[0] = *(const uint4*)(xs + boff + kb * 32);
        bb.q[1] = *(const uint4*)(xs + boff + kb * 32 + 16);
        acc = __builtin_amdgcn_wmma_f32_16x16x32_bf16(
            false, m[kb].v, false, bb.v, (short)0, acc, false, false);
      }
      const int n = n0 + pt * 16 + l15;
#pragma unroll
      for (int r = 0; r < 8; ++r) {
        const int o = ow + r + hi * 8;
        const size_t idx = ((size_t)b * 128 + o) * NPIX + n;
        out[idx] = acc[r] + cv[r] + x[idx];   // exact f32 residual
      }
    }
  }
}

// ---------------------------------------------------------------------------
extern "C" void kernel_launch(void* const* d_in, const int* in_sizes, int n_in,
                              void* d_out, int out_size, void* d_ws, size_t ws_size,
                              hipStream_t stream)
{
  const float* x  = (const float*)d_in[0];
  const float* Wq = (const float*)d_in[1];
  const float* bq = (const float*)d_in[2];
  const float* Wk = (const float*)d_in[3];
  const float* bk = (const float*)d_in[4];
  const float* Wv = (const float*)d_in[5];
  const float* bv = (const float*)d_in[6];
  const float* Wo = (const float*)d_in[7];
  const float* bo = (const float*)d_in[8];
  const float* gm = (const float*)d_in[9];
  float* out = (float*)d_out;

  float* ws   = (float*)d_ws;
  float* pS   = ws;                                          // 8*32*16384 f32
  float* pSum = pS   + (size_t)BATCH * GRAM_WGS * 16384;     // 8*32*128
  float* S    = pSum + (size_t)BATCH * GRAM_WGS * 128;       // 8*16384
  float* ssum = S    + (size_t)BATCH * 16384;                // 8*128
  float* A    = ssum + (size_t)BATCH * 128;                  // 8*16384
  float* cvec = A    + (size_t)BATCH * 16384;                // 8*128
  unsigned short* Mbf = (unsigned short*)(cvec + BATCH * 128); // 8*16384 bf16

  k_gram<<<BATCH * GRAM_WGS, 256, 0, stream>>>(x, pS, pSum);
  k_reduce<<<(BATCH * 16384) / (256 * 4), 256, 0, stream>>>(pS, pSum, S, ssum);
  k_energy_softmax<<<BATCH * 8, 256, 0, stream>>>(S, ssum, Wq, bq, Wk, bk, A);
  k_fold<<<BATCH * 8, 256, 0, stream>>>(A, Wv, bv, Wo, bo, gm, Mbf, cvec);
  k_final<<<BATCH * FIN_WGS, 256, 0, stream>>>(x, Mbf, cvec, out);

  (void)in_sizes; (void)n_in; (void)out_size; (void)ws_size;
}
